// GroupViT1D_65506841198794
// MI455X (gfx1250) — matrix operand order repaired and checked by reading, not compiled, not run on hardware
//
#include <hip/hip_runtime.h>

// ---- types ---------------------------------------------------------------
typedef float  vf8   __attribute__((ext_vector_type(8)));
typedef __bf16 vbf16 __attribute__((ext_vector_type(16)));
typedef __bf16 vbf8  __attribute__((ext_vector_type(8)));
typedef unsigned u32x4 __attribute__((ext_vector_type(4)));
typedef int      i32x8 __attribute__((ext_vector_type(8)));
typedef int      i32x4 __attribute__((ext_vector_type(4)));

#define DEV __device__ __forceinline__

// float -> bf16, round-to-nearest-even
DEV __bf16 f2bf(float f) {
  unsigned u = __builtin_bit_cast(unsigned, f);
  u += 0x7fffu + ((u >> 16) & 1u);
  unsigned short h = (unsigned short)(u >> 16);
  return __builtin_bit_cast(__bf16, h);
}
DEV float bf2f(__bf16 h) {
  unsigned u = ((unsigned)__builtin_bit_cast(unsigned short, h)) << 16;
  return __builtin_bit_cast(float, u);
}

// A fragment (this code's contiguous k order, also used for B): lane hi=0 holds K {0..15},
// hi=1 holds K {16..31}; two contiguous 16B chunks of one row.
DEV vbf16 load_a(const __bf16* row, int k0, int hi) {
  vbf8 a0 = *(const vbf8*)(row + k0 + hi * 16);
  vbf8 a1 = *(const vbf8*)(row + k0 + hi * 16 + 8);
  return __builtin_shufflevector(a0, a1, 0, 1, 2, 3, 4, 5, 6, 7, 8, 9, 10, 11,
                                 12, 13, 14, 15);
}
// B fragment (32x16) from a row of an [N,K] row-major operand in global mem.
DEV vbf16 load_b(const __bf16* row, int k0, int hi) {
  const vbf8* p = (const vbf8*)(row + k0 + hi * 16);
  vbf8 b0 = p[0], b1 = p[1];
  return __builtin_shufflevector(b0, b1, 0, 1, 2, 3, 4, 5, 6, 7, 8, 9, 10, 11,
                                 12, 13, 14, 15);
}
DEV vf8 wmma_bf16(vbf16 a, vbf16 b, vf8 c) {
  return __builtin_amdgcn_wmma_f32_16x16x32_bf16(false, a, false, b, (short)0,
                                                 c, false, false);
}

// ---- TDM: DMA one 64x128 bf16 tile (row stride `row_stride` elems) to LDS.
// Pads 4 DWORDs after every 64 DWORDs (one row) -> LDS row stride 136 elems,
// which keeps 16B alignment and spreads the 16 columns across LDS banks.
DEV void tdm_load_tile(const __bf16* gsrc, unsigned lds_off, int row_stride) {
  unsigned long long ga = (unsigned long long)(const void*)gsrc;
  u32x4 g0;
  g0[0] = 1u;                                                 // count = 1
  g0[1] = lds_off;                                            // LDS byte addr
  g0[2] = (unsigned)ga;                                       // gaddr[31:0]
  g0[3] = (unsigned)((ga >> 32) & 0x01ffffffu) | (2u << 30);  // gaddr hi,type=2
  i32x8 g1;
  g1[0] = (1 << 16)          // data_size = 2 bytes
        | (1 << 20)          // pad_enable
        | (5 << 22)          // pad_interval: 64 DWORDs (one row)
        | (3 << 25);         // pad_amount: 4 DWORDs
  g1[1] = (128 << 16);       // tensor_dim0 = 128
  g1[2] = (64 << 16);        // tensor_dim1 = 64
  g1[3] = (128 << 16);       // tile_dim0 = 128
  g1[4] = 64;                // tile_dim1 = 64, tile_dim2 = 0
  g1[5] = row_stride;        // tensor_dim0_stride
  g1[6] = 0;
  g1[7] = 0;
  i32x4 z4 = {0, 0, 0, 0};
#if defined(__clang_major__) && (__clang_major__ >= 23)
  i32x8 z8 = {0, 0, 0, 0, 0, 0, 0, 0};
  __builtin_amdgcn_tensor_load_to_lds(g0, g1, z4, z4, z8, 0);
#else
  __builtin_amdgcn_tensor_load_to_lds(g0, g1, z4, z4, 0);
#endif
}

// B fragment from the padded LDS tile (ds_load path).
DEV vbf16 load_b_lds(const __bf16 (&buf)[64][136], int col, int kl, int hi) {
  const __bf16* p = &buf[col][kl + hi * 16];
  vbf8 b0 = *(const vbf8*)p;
  vbf8 b1 = *(const vbf8*)(p + 8);
  return __builtin_shufflevector(b0, b1, 0, 1, 2, 3, 4, 5, 6, 7, 8, 9, 10, 11,
                                 12, 13, 14, 15);
}

// ---- fp32 -> bf16 bulk conversion (8 elems / thread) ---------------------
__global__ __launch_bounds__(256)
void cvt_kernel(const float* __restrict__ src, __bf16* __restrict__ dst,
                int n8) {
  int i = blockIdx.x * 256 + threadIdx.x;
  if (i >= n8) return;
  const float4* p = (const float4*)src + 2 * (size_t)i;
  float4 x = p[0], y = p[1];
  vbf8 o;
  o[0] = f2bf(x.x); o[1] = f2bf(x.y); o[2] = f2bf(x.z); o[3] = f2bf(x.w);
  o[4] = f2bf(y.x); o[5] = f2bf(y.y); o[6] = f2bf(y.z); o[7] = f2bf(y.w);
  ((vbf8*)dst)[i] = o;
}

// ---- GEMM: Out[T,N] = X[T,K] @ W[N,K]^T + bias (all-bf16, f32 acc) -------
// Block = 128 threads (4 waves stacked in M => 128 rows x 64 cols).
// B tile (64 x 128 per chunk) is staged to LDS by the Tensor Data Mover,
// double-buffered on TENSORcnt; A streams through global_load_b128.
template <bool OUT_F32>
__global__ __launch_bounds__(128)
void gemm_bf16(const __bf16* __restrict__ X, const __bf16* __restrict__ W,
               const float* __restrict__ bias, void* __restrict__ OutP, int K,
               int N) {
  __shared__ __bf16 Bs[2][64][136];  // padded rows (TDM pad feature)

  const int lane = threadIdx.x & 31;
  const int wave = threadIdx.x >> 5;
  const int nn = lane & 15;
  const int hi = lane >> 4;
  const int m0 = blockIdx.x * 128 + wave * 32;
  const int n0 = blockIdx.y * 64;

  const __bf16* ar0 = X + (size_t)(m0 + nn) * K;
  const __bf16* ar1 = X + (size_t)(m0 + 16 + nn) * K;
  const __bf16* wbase = W + (size_t)n0 * K;

  const int nchunks = K >> 7;  // K/128 (K % 128 == 0)
  if (wave == 0)
    tdm_load_tile(wbase, (unsigned)(unsigned long long)&Bs[0][0][0], K);

  vf8 acc[2][4] = {};

  for (int kc = 0; kc < nchunks; ++kc) {
    if (wave == 0) {
      if (kc + 1 < nchunks) {
        tdm_load_tile(wbase + (kc + 1) * 128,
                      (unsigned)(unsigned long long)&Bs[(kc + 1) & 1][0][0], K);
        __builtin_amdgcn_s_wait_tensorcnt(1);  // chunk kc complete (in-order)
      } else {
        __builtin_amdgcn_s_wait_tensorcnt(0);
      }
    }
    __syncthreads();

    const int kbase = kc << 7;
    __builtin_prefetch(ar0 + kbase + 256, 0, 0);  // global_prefetch_b8
    // A fragments for all 4 k-steps of this chunk (one load clause)
    vbf16 A0[4], A1[4];
#pragma unroll
    for (int s = 0; s < 4; ++s) {
      A0[s] = load_a(ar0, kbase + s * 32, hi);
      A1[s] = load_a(ar1, kbase + s * 32, hi);
    }
    const int bb = kc & 1;
#pragma unroll
    for (int s = 0; s < 4; ++s) {
      vbf16 b0 = load_b_lds(Bs[bb], nn, s * 32, hi);
      vbf16 b1 = load_b_lds(Bs[bb], 16 + nn, s * 32, hi);
      vbf16 b2 = load_b_lds(Bs[bb], 32 + nn, s * 32, hi);
      vbf16 b3 = load_b_lds(Bs[bb], 48 + nn, s * 32, hi);
      acc[0][0] = wmma_bf16(A0[s], b0, acc[0][0]);
      acc[1][0] = wmma_bf16(A1[s], b0, acc[1][0]);
      acc[0][1] = wmma_bf16(A0[s], b1, acc[0][1]);
      acc[1][1] = wmma_bf16(A1[s], b1, acc[1][1]);
      acc[0][2] = wmma_bf16(A0[s], b2, acc[0][2]);
      acc[1][2] = wmma_bf16(A1[s], b2, acc[1][2]);
      acc[0][3] = wmma_bf16(A0[s], b3, acc[0][3]);
      acc[1][3] = wmma_bf16(A1[s], b3, acc[1][3]);
    }
    __syncthreads();
  }

#pragma unroll
  for (int h = 0; h < 2; ++h)
#pragma unroll
    for (int t = 0; t < 4; ++t) {
      const int nc = n0 + t * 16 + nn;
      const float bia = bias[nc];
#pragma unroll
      for (int r = 0; r < 8; ++r) {
        const int mg = m0 + h * 16 + hi * 8 + r;  // C/D layout: row = hi*8+r
        const float val = acc[h][t][r] + bia;
        if (OUT_F32)
          ((float*)OutP)[(size_t)mg * N + nc] = val;
        else
          ((__bf16*)OutP)[(size_t)mg * N + nc] = f2bf(val);
      }
    }
}

// ---- logits + hard assignment -------------------------------------------
// Per wave: all 64 group rows x 16 tokens; argmax over rows in-register.
struct AFrags {
  vbf16 b, a0, a1, a2, a3;
};
DEV void load_afrags(AFrags& f, const __bf16* qb, const __bf16* krow, int k0,
                     int nn, int hi) {
  f.b = load_b(krow, k0, hi);
  f.a0 = load_a(qb + (size_t)(0 * 16 + nn) * 768, k0, hi);
  f.a1 = load_a(qb + (size_t)(1 * 16 + nn) * 768, k0, hi);
  f.a2 = load_a(qb + (size_t)(2 * 16 + nn) * 768, k0, hi);
  f.a3 = load_a(qb + (size_t)(3 * 16 + nn) * 768, k0, hi);
}
DEV void do_awmma(const AFrags& f, vf8 acc[4]) {
  acc[0] = wmma_bf16(f.a0, f.b, acc[0]);
  acc[1] = wmma_bf16(f.a1, f.b, acc[1]);
  acc[2] = wmma_bf16(f.a2, f.b, acc[2]);
  acc[3] = wmma_bf16(f.a3, f.b, acc[3]);
}

__global__ __launch_bounds__(256)
void assign_kernel(const __bf16* __restrict__ q, const __bf16* __restrict__ km,
                   int* __restrict__ assign) {
  const int lane = threadIdx.x & 31;
  const int wave = threadIdx.x >> 5;
  const int nn = lane & 15;
  const int hi = lane >> 4;
  const int b = blockIdx.y;
  const int s0 = (blockIdx.x * 8 + wave) * 16;

  const __bf16* qb = q + (size_t)b * 64 * 768;
  const __bf16* krow = km + ((size_t)b * 4096 + s0 + nn) * 768;

  vf8 acc[4] = {};
  AFrags f0, f1;
  load_afrags(f0, qb, krow, 0, nn, hi);
  for (int k0 = 0; k0 < 768; k0 += 64) {
    __builtin_prefetch(krow + k0 + 128, 0, 0);
    load_afrags(f1, qb, krow, k0 + 32, nn, hi);
    do_awmma(f0, acc);
    if (k0 + 64 < 768) load_afrags(f0, qb, krow, k0 + 64, nn, hi);
    do_awmma(f1, acc);
  }

  // per-lane argmax over 4 tiles x 8 rows (row = t*16 + hi*8 + r)
  float bestv = -3.4e38f;
  int bestm = 0;
#pragma unroll
  for (int t = 0; t < 4; ++t)
#pragma unroll
    for (int r = 0; r < 8; ++r) {
      const float vv = acc[t][r];
      const int m = t * 16 + hi * 8 + r;
      if (vv > bestv) { bestv = vv; bestm = m; }
    }
  // merge the two half-tiles (lane ^ 16 holds the other 32 rows)
  const float ov = __shfl_xor(bestv, 16, 32);
  const int om = __shfl_xor(bestm, 16, 32);
  if (ov > bestv || (ov == bestv && om < bestm)) { bestv = ov; bestm = om; }
  if (hi == 0) assign[b * 4096 + s0 + nn] = bestm;
}

// ---- segmented mean: gout[b,n,:] = sum_{assign==n} v[b,s,:]/(cnt+1) ------
__global__ __launch_bounds__(256)
void group_reduce(const __bf16* __restrict__ v, const int* __restrict__ assign,
                  __bf16* __restrict__ gout) {
  const int n = blockIdx.x;
  const int b = blockIdx.y;
  const int tid = threadIdx.x;
  const __bf16* vb = v + (size_t)b * 4096 * 768;
  const int* ab = assign + b * 4096;

  float a0 = 0.f, a1 = 0.f, a2 = 0.f;
  int cnt = 0;
  for (int s = 0; s < 4096; ++s) {
    if (ab[s] == n) {  // block-uniform scalar branch
      ++cnt;
      const __bf16* row = vb + (size_t)s * 768;
      a0 += bf2f(row[tid]);
      a1 += bf2f(row[tid + 256]);
      a2 += bf2f(row[tid + 512]);
    }
  }
  const float sc = 1.0f / ((float)cnt + 1.0f);  // ASSIGN_EPS = 1
  __bf16* g = gout + ((size_t)b * 64 + n) * 768;
  g[tid] = f2bf(a0 * sc);
  g[tid + 256] = f2bf(a1 * sc);
  g[tid + 512] = f2bf(a2 * sc);
}

// ---- host launch ---------------------------------------------------------
extern "C" void kernel_launch(void* const* d_in, const int* in_sizes, int n_in,
                              void* d_out, int out_size, void* d_ws,
                              size_t ws_size, hipStream_t stream) {
  (void)in_sizes; (void)n_in; (void)out_size; (void)ws_size;
  const float* query = (const float*)d_in[0];  // [16,64,768]
  const float* keyt  = (const float*)d_in[1];  // [16,4096,768]
  const float* Wq = (const float*)d_in[2]; const float* bq = (const float*)d_in[3];
  const float* Wk = (const float*)d_in[4]; const float* bk = (const float*)d_in[5];
  const float* Wv = (const float*)d_in[6]; const float* bv = (const float*)d_in[7];
  const float* Wo = (const float*)d_in[8]; const float* bo = (const float*)d_in[9];

  char* ws = (char*)d_ws;
  size_t off = 0;
  auto take = [&](size_t bytes) {
    void* p = ws + off;
    off += (bytes + 255) & ~(size_t)255;
    return p;
  };
  const size_t NQ = (size_t)16 * 64 * 768;    // 786,432
  const size_t NS = (size_t)16 * 4096 * 768;  // 50,331,648
  const size_t NW = (size_t)768 * 768;        // 589,824

  __bf16* query_bf = (__bf16*)take(NQ * 2);
  __bf16* keyt_bf  = (__bf16*)take(NS * 2);
  __bf16* Wq_bf = (__bf16*)take(NW * 2);
  __bf16* Wk_bf = (__bf16*)take(NW * 2);
  __bf16* Wv_bf = (__bf16*)take(NW * 2);
  __bf16* Wo_bf = (__bf16*)take(NW * 2);
  __bf16* qb = (__bf16*)take(NQ * 2);
  __bf16* kb = (__bf16*)take(NS * 2);
  __bf16* vb = (__bf16*)take(NS * 2);
  int* asg = (int*)take((size_t)16 * 4096 * 4);
  __bf16* go = (__bf16*)take(NQ * 2);

  auto cvt = [&](const float* s, __bf16* d, size_t n) {
    int n8 = (int)(n / 8);
    cvt_kernel<<<(n8 + 255) / 256, 256, 0, stream>>>(s, d, n8);
  };
  // bulk fp32 -> bf16 (bandwidth-bound, one pass)
  cvt(query, query_bf, NQ);
  cvt(keyt, keyt_bf, NS);
  cvt(Wq, Wq_bf, NW);
  cvt(Wk, Wk_bf, NW);
  cvt(Wv, Wv_bf, NW);
  cvt(Wo, Wo_bf, NW);

  const dim3 gblk(128);
  // q/k/v projections (bf16 out): T=1024 -> grid.x 8, T=65536 -> 512
  gemm_bf16<false><<<dim3(8, 12), gblk, 0, stream>>>(query_bf, Wq_bf, bq, qb, 768, 768);
  gemm_bf16<false><<<dim3(512, 12), gblk, 0, stream>>>(keyt_bf, Wk_bf, bk, kb, 768, 768);
  gemm_bf16<false><<<dim3(512, 12), gblk, 0, stream>>>(keyt_bf, Wv_bf, bv, vb, 768, 768);
  // hard assignment via in-register argmax of q.k^T (scale>0 is monotone)
  assign_kernel<<<dim3(32, 16), 256, 0, stream>>>(qb, kb, asg);
  // segmented mean with +1 renorm, bf16 out
  group_reduce<<<dim3(64, 16), 256, 0, stream>>>(vb, asg, go);
  // output projection (fp32 out)
  gemm_bf16<true><<<dim3(8, 12), gblk, 0, stream>>>(go, Wo_bf, bo, d_out, 768, 768);
}
